// MultiHeadAttention_54099408060886
// MI455X (gfx1250) — matrix operand order repaired and checked by reading, not compiled, not run on hardware
//
#include <hip/hip_runtime.h>

#define DEV __device__ __forceinline__

typedef __attribute__((ext_vector_type(16))) __bf16 v16bf;
typedef __attribute__((ext_vector_type(8)))  float  v8f;

constexpr int Bb = 4, Ss = 2048, Dd = 1024, Hh = 16, DK = 64;
constexpr int Nn = Bb * Ss;                       // 8192 rows

#if __has_builtin(__builtin_amdgcn_global_load_async_to_lds_b128) && \
    __has_builtin(__builtin_amdgcn_s_wait_asynccnt)
#define HAVE_ASYNC_LDS 1
typedef int v4i __attribute__((vector_size(4 * sizeof(int))));
typedef __attribute__((address_space(1))) v4i gv4i;   // global
typedef __attribute__((address_space(3))) v4i lv4i;   // LDS
#else
#define HAVE_ASYNC_LDS 0
#endif

// ---------------------------------------------------------------------------
// WMMA fragment helpers (contiguous k = 16*half + s in both A and B, wave32)
// ---------------------------------------------------------------------------
struct AFrag { union { v16bf v; unsigned int u[8]; }; };

// A (16x32, M x K): lane&15 = row m, lane>>4 = half.
// VGPR v: K = 2v + 16*half. Pairs contiguous
// -> one 32-byte contiguous chunk per lane.
DEV AFrag load_a_frag(const __bf16* src, int ld, int lane) {
  AFrag f;
  const int m = lane & 15, half = lane >> 4;
  const __bf16* row = src + (size_t)m * ld + half * 16;
#pragma unroll
  for (int v = 0; v < 4; ++v) f.u[v]     = *(const unsigned int*)(row + 2 * v);
#pragma unroll
  for (int v = 0; v < 4; ++v) f.u[4 + v] = *(const unsigned int*)(row + 8 + 2 * v);
  return f;
}

// B (32x16, K x N) from "NT" storage: element (k, col) at src[col*ld + k].
// lane&15 = col; lanes 0-15 hold K=0..15, lanes 16-31 hold K=16..31.
DEV AFrag load_b_frag(const __bf16* src, int ld, int lane) {
  AFrag f;
  const int c = lane & 15, half = lane >> 4;
  const __bf16* col = src + (size_t)c * ld + half * 16;
#pragma unroll
  for (int v = 0; v < 8; ++v) f.u[v] = *(const unsigned int*)(col + 2 * v);
  return f;
}

DEV v8f wmma_bf16(const AFrag& a, const AFrag& b, v8f c) {
  // (neg_a, A, neg_b, B, c_mod, C, reuse_a, reuse_b)
  return __builtin_amdgcn_wmma_f32_16x16x32_bf16(false, a.v, false, b.v,
                                                 (short)0, c, false, false);
}

// ---------------------------------------------------------------------------
// 16-lane max reduction via DPP16 (pure VALU, co-executes with WMMA):
// quad xor1 (0xB1), quad xor2 (0x4E), row_half_mirror (0x141, valid because
// values are quad-uniform), row_mirror (0x140, valid because 8-group-uniform).
// ---------------------------------------------------------------------------
template <int CTRL>
DEV float maxswap(float x) {
  const int xi = __builtin_bit_cast(int, x);
  const int yi = __builtin_amdgcn_update_dpp(xi, xi, CTRL, 0xf, 0xf, true);
  return fmaxf(x, __builtin_bit_cast(float, yi));
}
DEV float rmax16(float x) {
  x = maxswap<0xB1>(x);
  x = maxswap<0x4E>(x);
  x = maxswap<0x141>(x);
  x = maxswap<0x140>(x);
  return x;
}

// ---------------------------------------------------------------------------
// f32 -> bf16 conversion, float4 -> 64-bit packed stores
// ---------------------------------------------------------------------------
__global__ __launch_bounds__(256) void cvt_f32_bf16(const float* __restrict__ in,
                                                    __bf16* __restrict__ out,
                                                    size_t n4) {
  size_t i = (size_t)blockIdx.x * blockDim.x + threadIdx.x;
  const size_t stride = (size_t)gridDim.x * blockDim.x;
  for (; i < n4; i += stride) {
    const float4 v = ((const float4*)in)[i];
    union { __bf16 h[4]; unsigned long long q; } u;
    u.h[0] = (__bf16)v.x; u.h[1] = (__bf16)v.y;
    u.h[2] = (__bf16)v.z; u.h[3] = (__bf16)v.w;
    ((unsigned long long*)out)[i] = u.q;
  }
}

// ---------------------------------------------------------------------------
// C[N,Dd] = X[N,Dd](bf16) @ W[Dd,Dd]^T (bf16) + bias(f32)
// block = 256 (8 waves: 2 row groups x 4 col groups); block tile 128x128.
// wave tile 64x32 (4 m-tiles x 2 j-tiles): 6 frag loads per 8 WMMAs.
// ---------------------------------------------------------------------------
template <bool OUT_BF16>
__global__ __launch_bounds__(256) void gemm_nt(const __bf16* __restrict__ X,
                                               const __bf16* __restrict__ W,
                                               const float* __restrict__ bias,
                                               void* __restrict__ out) {
  const int lane = threadIdx.x & 31;
  const int wave = threadIdx.x >> 5;
  const int rg = wave >> 2;                 // 0..1
  const int cg = wave & 3;                  // 0..3
  const int row0 = blockIdx.y * 128 + rg * 64;
  const int col0 = blockIdx.x * 128 + cg * 32;

  v8f acc[4][2] = {};
  for (int k = 0; k < Dd; k += 32) {
    AFrag a[4], b[2];
#pragma unroll
    for (int mt = 0; mt < 4; ++mt)
      a[mt] = load_a_frag(X + (size_t)(row0 + 16 * mt) * Dd + k, Dd, lane);
#pragma unroll
    for (int jt = 0; jt < 2; ++jt)
      b[jt] = load_b_frag(W + (size_t)(col0 + 16 * jt) * Dd + k, Dd, lane);
#pragma unroll
    for (int mt = 0; mt < 4; ++mt)
#pragma unroll
      for (int jt = 0; jt < 2; ++jt)
        acc[mt][jt] = wmma_bf16(a[mt], b[jt], acc[mt][jt]);
  }

  const int n = lane & 15, half = lane >> 4;
#pragma unroll
  for (int mt = 0; mt < 4; ++mt)
#pragma unroll
    for (int jt = 0; jt < 2; ++jt) {
      const int col = col0 + jt * 16 + n;
      const float bv = bias[col];
#pragma unroll
      for (int r = 0; r < 8; ++r) {
        const size_t row = (size_t)(row0 + mt * 16 + r + 8 * half);
        const float val = acc[mt][jt][r] + bv;
        if (OUT_BF16) ((__bf16*)out)[row * Dd + col] = (__bf16)val;
        else          ((float*)out)[row * Dd + col] = val;
      }
    }
}

// ---------------------------------------------------------------------------
// Flash attention: grid (Ss/64, Bb*Hh), block 128 (4 waves, 16 q-rows/wave)
// Row sums computed with an extra "P @ ones" WMMA accumulator (no shuffles);
// row max via DPP16. K tile staged with async global->LDS when available.
// ---------------------------------------------------------------------------
__global__ __launch_bounds__(128) void flash_attn_kernel(
    const __bf16* __restrict__ Q, const __bf16* __restrict__ Km,
    const __bf16* __restrict__ Vm, __bf16* __restrict__ Ctx) {
  __shared__ __bf16 sK[32][72];      // K chunk, row-major [kv][d]
  __shared__ __bf16 sVt[64][34];     // V chunk, transposed [d][kv]
  __shared__ __bf16 sP[4][16][34];   // per-wave softmax probs [m][kv]

  const int tid  = threadIdx.x;
  const int lane = tid & 31;
  const int wave = tid >> 5;
  const int half = lane >> 4;
  const int ln16 = lane & 15;

  const int b = blockIdx.y / Hh;
  const int h = blockIdx.y % Hh;
  const int q0 = blockIdx.x * 64 + wave * 16;

  const size_t rowQ = (size_t)(b * Ss + q0);
  const AFrag qa0 = load_a_frag(Q + rowQ * Dd + h * DK + 0,  Dd, lane);
  const AFrag qa1 = load_a_frag(Q + rowQ * Dd + h * DK + 32, Dd, lane);

  AFrag ones;                         // all-ones bf16 B fragment (row-sum trick)
#pragma unroll
  for (int v = 0; v < 8; ++v) ones.u[v] = 0x3F803F80u;

  v8f o[4] = {};
  v8f osum = {};                      // running softmax denominator per row
  float rowmax[8];
#pragma unroll
  for (int r = 0; r < 8; ++r) rowmax[r] = -1.0e30f;

  const int nst = tid >> 2;           // 0..31: kv row this thread stages
  const int d0  = (tid & 3) * 16;     // 16-wide d slice

  for (int kv0 = 0; kv0 < Ss; kv0 += 32) {
    // ---- cooperative staging: K row-major (async if available), V transposed
    const size_t gRow = (size_t)(b * Ss + kv0 + nst);
    const __bf16* kg = Km + gRow * Dd + h * DK + d0;
    const __bf16* vg = Vm + gRow * Dd + h * DK + d0;
#if HAVE_ASYNC_LDS
    __builtin_amdgcn_global_load_async_to_lds_b128((gv4i*)kg,
                                                   (lv4i*)&sK[nst][d0], 0, 0);
    __builtin_amdgcn_global_load_async_to_lds_b128((gv4i*)kg,
                                                   (lv4i*)&sK[nst][d0], 16, 0);
#else
#pragma unroll
    for (int i = 0; i < 8; ++i)
      *(unsigned int*)&sK[nst][d0 + 2 * i] = *(const unsigned int*)(kg + 2 * i);
#endif
#pragma unroll
    for (int i = 0; i < 16; ++i)
      sVt[d0 + i][nst] = vg[i];
#if HAVE_ASYNC_LDS
    __builtin_amdgcn_s_wait_asynccnt(0);
#endif
    __syncthreads();

    // ---- scores S = Q K^T * 1/sqrt(DK) : two 16x16 n-tiles ----
    v8f s0 = {}, s1 = {};
    {
      AFrag kb;
      kb = load_b_frag(&sK[0][0],   72, lane);  s0 = wmma_bf16(qa0, kb, s0);
      kb = load_b_frag(&sK[0][32],  72, lane);  s0 = wmma_bf16(qa1, kb, s0);
      kb = load_b_frag(&sK[16][0],  72, lane);  s1 = wmma_bf16(qa0, kb, s1);
      kb = load_b_frag(&sK[16][32], 72, lane);  s1 = wmma_bf16(qa1, kb, s1);
    }
#pragma unroll
    for (int r = 0; r < 8; ++r) { s0[r] *= 0.125f; s1[r] *= 0.125f; }

    // ---- online softmax update (rows live in 16-lane groups) ----
#pragma unroll
    for (int r = 0; r < 8; ++r) {
      const float cm = rmax16(fmaxf(s0[r], s1[r]));
      const float nm = fmaxf(rowmax[r], cm);
      const float alpha = __expf(rowmax[r] - nm);
      rowmax[r] = nm;
      s0[r] = __expf(s0[r] - nm);
      s1[r] = __expf(s1[r] - nm);
      o[0][r] *= alpha; o[1][r] *= alpha;
      o[2][r] *= alpha; o[3][r] *= alpha;
      osum[r] *= alpha;
    }

    // ---- C-layout f32 -> A-layout bf16 via per-wave LDS ----
#pragma unroll
    for (int r = 0; r < 8; ++r) {
      sP[wave][r + 8 * half][ln16]      = (__bf16)s0[r];
      sP[wave][r + 8 * half][16 + ln16] = (__bf16)s1[r];
    }
    const AFrag pa = load_a_frag(&sP[wave][0][0], 34, lane);

    // ---- O += P V (4 d-tiles) ; denom += P @ ones ----
    {
      AFrag vb;
      vb = load_b_frag(&sVt[0][0],  34, lane);  o[0] = wmma_bf16(pa, vb, o[0]);
      vb = load_b_frag(&sVt[16][0], 34, lane);  o[1] = wmma_bf16(pa, vb, o[1]);
      vb = load_b_frag(&sVt[32][0], 34, lane);  o[2] = wmma_bf16(pa, vb, o[2]);
      vb = load_b_frag(&sVt[48][0], 34, lane);  o[3] = wmma_bf16(pa, vb, o[3]);
      osum = wmma_bf16(pa, ones, osum);
    }
    __syncthreads();
  }

  // ---- normalize and emit merged-head ctx (bf16, [B,S,D]) ----
#pragma unroll
  for (int r = 0; r < 8; ++r) {
    const float inv = 1.0f / osum[r];
    const size_t row = (size_t)(b * Ss + q0 + r + 8 * half);
#pragma unroll
    for (int t = 0; t < 4; ++t)
      Ctx[row * Dd + h * DK + 16 * t + ln16] = (__bf16)(o[t][r] * inv);
  }
}

// ---------------------------------------------------------------------------
extern "C" void kernel_launch(void* const* d_in, const int* in_sizes, int n_in,
                              void* d_out, int out_size, void* d_ws, size_t ws_size,
                              hipStream_t stream) {
  (void)in_sizes; (void)n_in; (void)out_size; (void)ws_size;

  const float* query = (const float*)d_in[0];
  const float* key_  = (const float*)d_in[1];
  const float* value = (const float*)d_in[2];
  const float* Wq = (const float*)d_in[3];
  const float* bq = (const float*)d_in[4];
  const float* Wk = (const float*)d_in[5];
  const float* bk = (const float*)d_in[6];
  const float* Wv = (const float*)d_in[7];
  const float* bv = (const float*)d_in[8];
  const float* Wo = (const float*)d_in[9];
  const float* bo = (const float*)d_in[10];

  const size_t ND = (size_t)Nn * Dd;   // 8 Mi elements
  const size_t DD = (size_t)Dd * Dd;   // 1 Mi elements

  __bf16* Xq  = (__bf16*)d_ws;
  __bf16* Xk  = Xq  + ND;
  __bf16* Xv  = Xk  + ND;
  __bf16* Wqb = Xv  + ND;
  __bf16* Wkb = Wqb + DD;
  __bf16* Wvb = Wkb + DD;
  __bf16* Wob = Wvb + DD;
  __bf16* Qb  = Wob + DD;
  __bf16* Kb  = Qb  + ND;
  __bf16* Vb  = Kb  + ND;
  __bf16* Cb  = Vb  + ND;

  cvt_f32_bf16<<<1024, 256, 0, stream>>>(query, Xq, ND / 4);
  cvt_f32_bf16<<<1024, 256, 0, stream>>>(key_,  Xk, ND / 4);
  cvt_f32_bf16<<<1024, 256, 0, stream>>>(value, Xv, ND / 4);
  cvt_f32_bf16<<<256,  256, 0, stream>>>(Wq, Wqb, DD / 4);
  cvt_f32_bf16<<<256,  256, 0, stream>>>(Wk, Wkb, DD / 4);
  cvt_f32_bf16<<<256,  256, 0, stream>>>(Wv, Wvb, DD / 4);
  cvt_f32_bf16<<<256,  256, 0, stream>>>(Wo, Wob, DD / 4);

  const dim3 gg(Dd / 128, Nn / 128);
  gemm_nt<true><<<gg, 256, 0, stream>>>(Xq, Wqb, bq, Qb);
  gemm_nt<true><<<gg, 256, 0, stream>>>(Xk, Wkb, bk, Kb);
  gemm_nt<true><<<gg, 256, 0, stream>>>(Xv, Wvb, bv, Vb);

  flash_attn_kernel<<<dim3(Ss / 64, Bb * Hh), 128, 0, stream>>>(Qb, Kb, Vb, Cb);

  gemm_nt<false><<<gg, 256, 0, stream>>>(Cb, Wob, bo, d_out);
}
